// DeformConv2dBlock_70729521430882
// MI455X (gfx1250) — compile-verified, matrix operand order checked by reading, not run on hardware
//
#include <hip/hip_runtime.h>
#include <math.h>

typedef __attribute__((ext_vector_type(16))) _Float16 v16h;
typedef __attribute__((ext_vector_type(8)))  _Float16 v8h;
typedef __attribute__((ext_vector_type(8)))  float    v8f;

#define CB    4
#define CIN   64
#define HH    96
#define WW    96
#define OUTC  128
#define KK2   9
#define HW    (HH*WW)        // 9216
#define KTOT  (CIN*KK2)      // 576
#define MPIX  (CB*HW)        // 36864

#define LDS_STRIDE 584                       // 576 + 8 f16 pad; 1168B % 256 == 144
#define LDS_BYTES  (OUTC * LDS_STRIDE * 2)   // 149,504 B  (< 160 KB)

// ---------------------------------------------------------------------------
// Kernel A: fused offset-conv (18 ch) + modulator-conv (9 ch), f32.
// Weights (27 x 576 f32 = 62 KB) staged in LDS, broadcast-read per FMA.
// ---------------------------------------------------------------------------
__global__ __launch_bounds__(256)
void conv_offmod_kernel(const float* __restrict__ x,
                        const float* __restrict__ offw,
                        const float* __restrict__ offb,
                        const float* __restrict__ modw,
                        const float* __restrict__ modb,
                        float* __restrict__ off_out,   // (B,18,H,W)
                        float* __restrict__ mod_out)   // (B,9,H,W)
{
    __shared__ float lw[27 * KTOT];   // [oc][c][tap], oc 0..17 offset, 18..26 mod
    const int tid = threadIdx.x;
    for (int i = tid; i < 27 * KTOT; i += 256)
        lw[i] = (i < 18 * KTOT) ? offw[i] : modw[i - 18 * KTOT];
    __syncthreads();

    const int m  = blockIdx.x * 256 + tid;     // one pixel per thread (grid exact)
    const int b  = m / HW;
    const int hw = m % HW;
    const int oh = hw / WW;
    const int ow = hw % WW;

    float acc[27];
#pragma unroll
    for (int oc = 0; oc < 18; ++oc) acc[oc] = offb[oc];
#pragma unroll
    for (int oc = 0; oc < 9;  ++oc) acc[18 + oc] = modb[oc];

    const float* xb = x + (size_t)b * CIN * HW;
    for (int c = 0; c < CIN; ++c) {
        const float* xc = xb + (size_t)c * HW;
        const float* wc = lw + c * 9;
#pragma unroll
        for (int t = 0; t < 9; ++t) {
            const int iy = oh - 1 + t / 3;
            const int ix = ow - 1 + t % 3;
            if (iy < 0 || iy >= HH || ix < 0 || ix >= WW) continue;
            const float xv = xc[iy * WW + ix];
#pragma unroll
            for (int oc = 0; oc < 27; ++oc)
                acc[oc] = fmaf(xv, wc[oc * KTOT + t], acc[oc]);
        }
    }

#pragma unroll
    for (int oc = 0; oc < 18; ++oc)
        off_out[((size_t)b * 18 + oc) * HW + hw] = acc[oc];
#pragma unroll
    for (int oc = 0; oc < 9; ++oc) {
        const float s = acc[18 + oc];
        mod_out[((size_t)b * 9 + oc) * HW + hw] = 2.0f / (1.0f + __expf(-s));
    }
}

// ---------------------------------------------------------------------------
// Kernel B: deformable bilinear gather + modulation -> im2col A [M x 576] f16.
// k-index = k2*64 + c.  Threads map ow-fastest: offset/mod reads and the four
// x gathers are lane-coalesced.  Channels packed 8-at-a-time into v8h so the
// A-matrix writes are 16B global_store_b128.
// ---------------------------------------------------------------------------
__global__ __launch_bounds__(256)
void im2col_deform_kernel(const float* __restrict__ x,
                          const float* __restrict__ off,   // (B,18,H,W)
                          const float* __restrict__ mod,   // (B,9,H,W)
                          _Float16* __restrict__ Amat)     // [MPIX][KTOT]
{
    const int gid = blockIdx.x * 256 + threadIdx.x;        // grid exact: M*9
    const int ow = gid % WW;
    const int t1 = gid / WW;
    const int oh = t1 % HH;
    const int t2 = t1 / HH;
    const int k2 = t2 % KK2;
    const int b  = t2 / KK2;

    const int hw = oh * WW + ow;
    const int m  = b * HW + hw;

    const float dy = off[((size_t)b * 18 + 2 * k2 + 0) * HW + hw];
    const float dx = off[((size_t)b * 18 + 2 * k2 + 1) * HW + hw];
    const float mk = mod[((size_t)b * 9 + k2) * HW + hw];

    const float py = (float)(oh - 1 + k2 / 3) + dy;
    const float px = (float)(ow - 1 + k2 % 3) + dx;
    const float fy = floorf(py);
    const float fx = floorf(px);
    const float ly = py - fy;
    const float lx = px - fx;
    const int y0 = (int)fy;
    const int x0 = (int)fx;
    const int y1 = y0 + 1;
    const int x1 = x0 + 1;

    const float vy0 = (y0 >= 0 && y0 < HH) ? 1.0f : 0.0f;
    const float vy1 = (y1 >= 0 && y1 < HH) ? 1.0f : 0.0f;
    const float vx0 = (x0 >= 0 && x0 < WW) ? 1.0f : 0.0f;
    const float vx1 = (x1 >= 0 && x1 < WW) ? 1.0f : 0.0f;

    const int cy0 = min(max(y0, 0), HH - 1);
    const int cy1 = min(max(y1, 0), HH - 1);
    const int cx0 = min(max(x0, 0), WW - 1);
    const int cx1 = min(max(x1, 0), WW - 1);

    const int i00 = cy0 * WW + cx0;
    const int i01 = cy0 * WW + cx1;
    const int i10 = cy1 * WW + cx0;
    const int i11 = cy1 * WW + cx1;

    const float w00 = (1.0f - ly) * (1.0f - lx) * vy0 * vx0 * mk;
    const float w01 = (1.0f - ly) * lx          * vy0 * vx1 * mk;
    const float w10 = ly * (1.0f - lx)          * vy1 * vx0 * mk;
    const float w11 = ly * lx                   * vy1 * vx1 * mk;

    const float* xb = x + (size_t)b * CIN * HW;
    _Float16* arow  = Amat + (size_t)m * KTOT + k2 * CIN;

    for (int c0 = 0; c0 < CIN; c0 += 8) {
        v8h pack;
#pragma unroll
        for (int j = 0; j < 8; ++j) {
            const float* xc = xb + (size_t)(c0 + j) * HW;
            const float v = w00 * xc[i00] + w01 * xc[i01] +
                            w10 * xc[i10] + w11 * xc[i11];
            pack[j] = (_Float16)v;
        }
        *(v8h*)(arow + c0) = pack;       // 16B aligned: 1152*m + 128*k2 + 2*c0
    }
}

// ---------------------------------------------------------------------------
// Kernel C: pack deform weight (O,C,3,3) f32 -> Wt[O][k2*64+c] f16 (N x K).
// ---------------------------------------------------------------------------
__global__ __launch_bounds__(256)
void pack_weight_kernel(const float* __restrict__ w, _Float16* __restrict__ wt)
{
    const int gid = blockIdx.x * 256 + threadIdx.x;   // grid exact: 128*576
    const int o = gid / KTOT;
    const int k = gid % KTOT;
    const int k2 = k / CIN;
    const int c  = k % CIN;
    wt[gid] = (_Float16)w[(size_t)o * KTOT + c * KK2 + k2];
}

// ---------------------------------------------------------------------------
// Kernel D: WMMA GEMM.  D[M=36864, N=128] = A[M,576] * Wt^T  (f16 in, f32 acc)
// Full weight matrix staged once per block in LDS (144KB, padded rows ->
// bank-spread ds_load_b128), shared by 8 waves x 2 m-tiles each.
// Per k-step: burst-load all 8 B fragments into distinct registers (one
// s_wait_dscnt for 16 clause'd ds_loads), then 8 back-to-back WMMAs.
// A fragment prefetched from global one k-step ahead.  k-loop kept rolled.
// ---------------------------------------------------------------------------
__global__ __launch_bounds__(256)
void gemm_wmma_kernel(const _Float16* __restrict__ Amat,
                      const _Float16* __restrict__ Wt,
                      float* __restrict__ out)
{
    extern __shared__ _Float16 lw[];               // [OUTC][LDS_STRIDE]

    const int tid   = threadIdx.x;
    const int lane  = tid & 31;
    const int wave  = tid >> 5;
    const int row   = lane & 15;
    const int khalf = lane >> 4;                   // kb = khalf*8

    // ---- stage Wt into LDS: 9216 chunks of 8 f16, 36 per thread ----
    for (int id = tid; id < OUTC * (KTOT / 8); id += 256) {
        const int n  = id / (KTOT / 8);
        const int kc = id - n * (KTOT / 8);
        *(v8h*)(lw + n * LDS_STRIDE + kc * 8) =
            *(const v8h*)(Wt + (size_t)n * KTOT + kc * 8);
    }
    __syncthreads();

    const v8f vzero = {0.f, 0.f, 0.f, 0.f, 0.f, 0.f, 0.f, 0.f};
    const _Float16* lrow = lw + row * LDS_STRIDE + khalf * 8;   // n-tile 0 base

#pragma unroll 1
    for (int it = 0; it < 2; ++it) {               // 2 m-tiles per wave
        const int mt = (blockIdx.x * 8 + wave) * 2 + it;
        const int m0 = mt * 16;

        const _Float16* arow =
            Amat + (size_t)(m0 + row) * KTOT + khalf * 8;

        v8f acc[8];
#pragma unroll
        for (int nt = 0; nt < 8; ++nt) acc[nt] = vzero;

        // software-pipelined A feed (global, one k-step ahead)
        v8h alo = *(const v8h*)(arow);
        v8h ahi = *(const v8h*)(arow + 16);

#pragma unroll 1
        for (int kk = 0; kk < KTOT; kk += 32) {
            const v16h afrag = __builtin_shufflevector(
                alo, ahi, 0, 1, 2, 3, 4, 5, 6, 7, 8, 9, 10, 11, 12, 13, 14, 15);

            if (kk + 32 < KTOT) {                  // prefetch next A fragment
                alo = *(const v8h*)(arow + kk + 32);
                ahi = *(const v8h*)(arow + kk + 48);
            }

            // burst-load all 8 B fragments (16 ds_load_b128, one wait)
            v8h blo[8], bhi[8];
#pragma unroll
            for (int nt = 0; nt < 8; ++nt) {
                const _Float16* brow = lrow + nt * 16 * LDS_STRIDE + kk;
                blo[nt] = *(const v8h*)(brow);
                bhi[nt] = *(const v8h*)(brow + 16);
            }

            // 8 back-to-back WMMAs (same A, independent accumulators)
#pragma unroll
            for (int nt = 0; nt < 8; ++nt) {
                const v16h bfrag = __builtin_shufflevector(
                    blo[nt], bhi[nt],
                    0, 1, 2, 3, 4, 5, 6, 7, 8, 9, 10, 11, 12, 13, 14, 15);
                acc[nt] = __builtin_amdgcn_wmma_f32_16x16x32_f16(
                    false, afrag, false, bfrag, (short)0, acc[nt], false, false);
            }
        }

        // ---- D scatter -> out (B,O,H,W), two b128 stores per n-tile ----
        const int b   = m0 / HW;                   // tiles never straddle batches
        const int hw0 = (m0 + khalf * 8) % HW;     // 8 consecutive hw per lane
#pragma unroll
        for (int nt = 0; nt < 8; ++nt) {
            const int n = nt * 16 + row;
            float* obase = out + ((size_t)b * OUTC + n) * HW + hw0;
            float4 lo = make_float4(acc[nt][0], acc[nt][1], acc[nt][2], acc[nt][3]);
            float4 hi = make_float4(acc[nt][4], acc[nt][5], acc[nt][6], acc[nt][7]);
            *(float4*)(obase)     = lo;            // 32B-aligned (hw0 % 8 == 0)
            *(float4*)(obase + 4) = hi;
        }
    }
}

// ---------------------------------------------------------------------------
// Launch: conv(offsets+mod) -> weight pack -> deform im2col -> WMMA GEMM.
// Workspace (all 256B-aligned):
//   [0)          off      4*18*9216 f32  = 2,654,208 B
//   [+2654208)   mod      4* 9*9216 f32  = 1,327,104 B
//   [+3981312)   Wt f16   128*576        =   147,456 B
//   [+4128768)   A  f16   36864*576      = 42,467,328 B     (total ~46.6 MB)
// ---------------------------------------------------------------------------
extern "C" void kernel_launch(void* const* d_in, const int* in_sizes, int n_in,
                              void* d_out, int out_size, void* d_ws, size_t ws_size,
                              hipStream_t stream)
{
    (void)in_sizes; (void)n_in; (void)out_size; (void)ws_size;

    const float* x    = (const float*)d_in[0];
    const float* offw = (const float*)d_in[1];
    const float* offb = (const float*)d_in[2];
    const float* modw = (const float*)d_in[3];
    const float* modb = (const float*)d_in[4];
    const float* dw   = (const float*)d_in[5];
    float* out = (float*)d_out;

    char* ws = (char*)d_ws;
    float*    ws_off = (float*)(ws);
    float*    ws_mod = (float*)(ws + 2654208);
    _Float16* ws_wt  = (_Float16*)(ws + 3981312);
    _Float16* ws_A   = (_Float16*)(ws + 4128768);

    // allow >64KB dynamic LDS for the GEMM (capture-safe, deterministic)
    hipFuncSetAttribute(reinterpret_cast<const void*>(gemm_wmma_kernel),
                        hipFuncAttributeMaxDynamicSharedMemorySize, LDS_BYTES);

    conv_offmod_kernel<<<MPIX / 256, 256, 0, stream>>>(
        x, offw, offb, modw, modb, ws_off, ws_mod);

    pack_weight_kernel<<<(OUTC * KTOT) / 256, 256, 0, stream>>>(dw, ws_wt);

    im2col_deform_kernel<<<(MPIX * KK2) / 256, 256, 0, stream>>>(
        x, ws_off, ws_mod, ws_A);

    // 2304 m-tiles / (8 waves * 2 tiles) = 144 blocks
    gemm_wmma_kernel<<<144, 256, LDS_BYTES, stream>>>(ws_A, ws_wt, out);
}
